// MixHop_76579266888074
// MI455X (gfx1250) — matrix operand order repaired and checked by reading, not compiled, not run on hardware
//
#include <hip/hip_runtime.h>
#include <hip/hip_bf16.h>

typedef __attribute__((ext_vector_type(16))) _Float16 v16h;
typedef __attribute__((ext_vector_type(4)))  _Float16 v4h;
typedef __attribute__((ext_vector_type(8)))  float    v8f;

// ---------------------------------------------------------------------------
// Degree via L2 atomics (counts incoming edges; self loop added in k_dinv)
// ---------------------------------------------------------------------------
__global__ void k_degree(const int* __restrict__ dst, float* __restrict__ deg, int E) {
  int e = blockIdx.x * blockDim.x + threadIdx.x;
  if (e < E) atomicAdd(&deg[dst[e]], 1.0f);
}

__global__ void k_dinv(float* __restrict__ deg, int N) {
  int i = blockIdx.x * blockDim.x + threadIdx.x;
  if (i < N) deg[i] = rsqrtf(deg[i] + 1.0f);  // +1 = self loop; always > 0
}

// ---------------------------------------------------------------------------
// out[i,:] = dinv[i]^2 * x[i,:]  (self-loop term; also initializes the buffer
// so the edge kernel can scatter-add without a separate memset)
// ---------------------------------------------------------------------------
__global__ void k_self(const float4* __restrict__ x4, const float* __restrict__ dinv,
                       float4* __restrict__ out4, int N, int F4) {
  int i = blockIdx.x * blockDim.x + threadIdx.x;
  if (i >= N * F4) return;
  float d = dinv[i / F4];
  d = d * d;
  float4 v = x4[i];
  out4[i] = make_float4(d * v.x, d * v.y, d * v.z, d * v.w);
}

// ---------------------------------------------------------------------------
// Edge-parallel SpMM: one wave32 per edge, float4 gathers from the source row
// (L2-resident), f32 atomic scatter-add into the destination row.
// ---------------------------------------------------------------------------
__global__ void k_spmm(const int* __restrict__ src, const int* __restrict__ dst,
                       const float* __restrict__ dinv, const float* __restrict__ x,
                       float* __restrict__ out, int E, int F) {
  int e    = (blockIdx.x * blockDim.x + threadIdx.x) >> 5;
  int lane = threadIdx.x & 31;
  if (e >= E) return;
  if (e + 2048 < E) __builtin_prefetch(&src[e + 2048], 0, 0);  // global_prefetch_b8
  int s = src[e], d = dst[e];
  float c = dinv[s] * dinv[d];
  const float4* xr = (const float4*)(x + (size_t)s * F);
  float* orow = out + (size_t)d * F;
  for (int f = lane * 4; f < F; f += 128) {
    float4 v = xr[f >> 2];
    atomicAdd(&orow[f + 0], c * v.x);
    atomicAdd(&orow[f + 1], c * v.y);
    atomicAdd(&orow[f + 2], c * v.z);
    atomicAdd(&orow[f + 3], c * v.w);
  }
}

// ---------------------------------------------------------------------------
// Pack 16 fp32 -> f16 A fragment per the CDNA5 16-bit A-matrix lane layout:
// element i<8  -> K = kbase + i        (floats p[0..7])
// element i>=8 -> K = kbase + 16 + i-8 (floats p[16..23])
// ---------------------------------------------------------------------------
__device__ __forceinline__ v16h packA(const float* __restrict__ p) {
  float4 f0 = *(const float4*)(p);
  float4 f1 = *(const float4*)(p + 4);
  float4 f2 = *(const float4*)(p + 16);
  float4 f3 = *(const float4*)(p + 20);
  v16h a;
  a[0]=(_Float16)f0.x;  a[1]=(_Float16)f0.y;  a[2]=(_Float16)f0.z;  a[3]=(_Float16)f0.w;
  a[4]=(_Float16)f1.x;  a[5]=(_Float16)f1.y;  a[6]=(_Float16)f1.z;  a[7]=(_Float16)f1.w;
  a[8]=(_Float16)f2.x;  a[9]=(_Float16)f2.y;  a[10]=(_Float16)f2.z; a[11]=(_Float16)f2.w;
  a[12]=(_Float16)f3.x; a[13]=(_Float16)f3.y; a[14]=(_Float16)f3.z; a[15]=(_Float16)f3.w;
  return a;
}

__device__ __forceinline__ v16h loadB(const _Float16* __restrict__ bp) {
  v16h b;
  #pragma unroll
  for (int i = 0; i < 16; ++i) b[i] = bp[i < 8 ? i : i + 8];
  return b;
}

// ---------------------------------------------------------------------------
// WMMA GEMM: C[M,Nout] = A[M,K] @ W[K,Nout] + bias (optional ReLU).
// Block = 256 threads (8 waves), block tile 128x16, K-step 64 (2 WMMA/step).
// OOB rows/cols handled by ADDRESS CLAMPING, not zero-fill: clamped rows and
// columns produce garbage only in outputs the epilogue never stores, so the
// whole K-loop is branch-free.
// Staging: thread t owns column n=t&15, K quad k=(t>>4)*4 -> 4 clause-able
// global b32 loads, 2 v_cvt_pk_f16_f32, 1 ds_store_b64.
// ---------------------------------------------------------------------------
__global__ void k_gemm_wmma(const float* __restrict__ A, int lda,
                            const float* __restrict__ W, int Nout,
                            const float* __restrict__ bias,
                            float* __restrict__ C, int ldc,
                            int M, int K, int relu) {
  __shared__ _Float16 Bt[16 * 72];  // [n][k 0..63], stride 72 -> conflict-free b128 reads

  const int tid  = threadIdx.x;
  const int wave = tid >> 5;
  const int lane = tid & 31;
  const int m0   = blockIdx.x * 128 + wave * 16;
  const int n0   = blockIdx.y * 16;

  int row = m0 + (lane & 15);                 // A-matrix: M = lane&15
  row = row < M ? row : M - 1;                // clamp (garbage rows never stored)
  const int kbase = (lane >> 4) * 8;          // lane<16 -> K 0..7/16..23, else 8..15/24..31
  const int nloc  = lane & 15;                // B-matrix: column = lane&15
  const int khalf = (lane >> 4) * 8;          // B-matrix: lane<16 -> K 0..7/16..23, else 8..15/24..31

  const float*    ap0 = A + (size_t)row * lda + kbase;
  const _Float16* bp  = &Bt[nloc * 72 + khalf];

  // Staging assignment (k0-invariant): column + K-quad per thread
  const int sn = tid & 15;
  const int sk = (tid >> 4) * 4;              // 0,4,...,60
  int snc = n0 + sn;
  snc = snc < Nout ? snc : Nout - 1;          // clamp (garbage cols never stored)
  const float* wcol = W + (size_t)sk * Nout + snc;   // advance by 64*Nout per K-step
  _Float16* sdst = &Bt[sn * 72 + sk];

  v8f acc = {};
  for (int k0 = 0; k0 < K; k0 += 64) {        // K is 256 or 384 -> always full steps
    __syncthreads();
    {
      const float* wp = wcol + (size_t)k0 * Nout;
      float w0 = wp[0];
      float w1 = wp[(size_t)Nout];
      float w2 = wp[(size_t)2 * Nout];
      float w3 = wp[(size_t)3 * Nout];
      v4h hv;
      hv[0] = (_Float16)w0; hv[1] = (_Float16)w1;
      hv[2] = (_Float16)w2; hv[3] = (_Float16)w3;
      *(v4h*)sdst = hv;                       // ds_store_b64
    }
    __syncthreads();

    const float* ap = ap0 + k0;
    v16h a0 = packA(ap);                      // K local 0..31
    v16h a1 = packA(ap + 32);                 // K local 32..63
    v16h b0 = loadB(bp);                      // Bt[n][khalf..+7, khalf+16..+23]
    v16h b1 = loadB(bp + 32);                 // Bt[n][32+khalf..+7, 48+khalf..+7]

    acc = __builtin_amdgcn_wmma_f32_16x16x32_f16(false, a0, false, b0,
                                                 (short)0, acc, false, false);
    acc = __builtin_amdgcn_wmma_f32_16x16x32_f16(false, a1, false, b1,
                                                 (short)0, acc, false, false);
  }

  // Epilogue: C/D layout -> VGPR r: lanes0-15 M=r, lanes16-31 M=8+r; N=lane&15
  const int col   = n0 + nloc;
  const int rbase = m0 + (lane >> 4) * 8;
  if (col < Nout) {
    float bv = bias[col];
    #pragma unroll
    for (int r = 0; r < 8; ++r) {
      int rr = rbase + r;
      if (rr < M) {
        float v = acc[r] + bv;
        if (relu) v = fmaxf(v, 0.0f);
        C[(size_t)rr * ldc + col] = v;
      }
    }
  }
}

// ---------------------------------------------------------------------------
// In-place row-wise log_softmax, one wave per row (Ccols <= 128)
// ---------------------------------------------------------------------------
__global__ void k_logsoftmax(float* __restrict__ out, int N, int Ccols) {
  int wave = threadIdx.x >> 5;
  int lane = threadIdx.x & 31;
  int row  = blockIdx.x * (blockDim.x >> 5) + wave;
  if (row >= N) return;
  float* r = out + (size_t)row * Ccols;
  float vals[4];
  int cnt = 0;
  float m = -3.4e38f;
  for (int j = lane; j < Ccols; j += 32) { float v = r[j]; vals[cnt++] = v; m = fmaxf(m, v); }
  for (int o = 16; o; o >>= 1) m = fmaxf(m, __shfl_xor(m, o, 32));
  float s = 0.0f;
  cnt = 0;
  for (int j = lane; j < Ccols; j += 32) s += expf(vals[cnt++] - m);
  for (int o = 16; o; o >>= 1) s += __shfl_xor(s, o, 32);
  float ls = logf(s);
  cnt = 0;
  for (int j = lane; j < Ccols; j += 32) r[j] = vals[cnt++] - m - ls;
}

// ---------------------------------------------------------------------------
extern "C" void kernel_launch(void* const* d_in, const int* in_sizes, int n_in,
                              void* d_out, int out_size, void* d_ws, size_t ws_size,
                              hipStream_t stream) {
  const int F  = 256, H = 128, Cc = 40;
  const int H3 = 3 * H;    // 384
  const int C3 = 3 * Cc;   // 120
  const int N  = in_sizes[0] / F;
  const int E  = in_sizes[1] / 2;

  const float* x    = (const float*)d_in[0];
  const int*   esrc = (const int*)d_in[1];
  const int*   edst = esrc + E;
  const float* W1_0 = (const float*)d_in[2];
  const float* W1_1 = (const float*)d_in[3];
  const float* W1_2 = (const float*)d_in[4];
  const float* b1   = (const float*)d_in[5];
  const float* W2_0 = (const float*)d_in[6];
  const float* W2_1 = (const float*)d_in[7];
  const float* W2_2 = (const float*)d_in[8];
  const float* b2   = (const float*)d_in[9];
  float*       outv = (float*)d_out;

  // Workspace layout (floats): dinv | p1(N*F) | p2(N*F) | h(N*H3) | q1(N*H3)
  // q2 reuses the dead p1/p2 region (N*2F >= N*H3). Total ~N*1281 floats.
  float* wsf  = (float*)d_ws;
  float* dinv = wsf;
  float* p1   = dinv + N;
  float* p2   = p1 + (size_t)N * F;
  float* h    = p2 + (size_t)N * F;
  float* q1   = h  + (size_t)N * H3;
  float* q2   = p1;  // reuse

  // ---- normalization: deg -> dinv
  hipMemsetAsync(dinv, 0, (size_t)N * sizeof(float), stream);
  k_degree<<<(E + 255) / 256, 256, 0, stream>>>(edst, dinv, E);
  k_dinv<<<(N + 255) / 256, 256, 0, stream>>>(dinv, N);

  // ---- layer 1 propagates: p1 = P(x), p2 = P(p1)
  int t4a = N * (F / 4);
  k_self<<<(t4a + 255) / 256, 256, 0, stream>>>((const float4*)x, dinv, (float4*)p1, N, F / 4);
  k_spmm<<<(E + 7) / 8, 256, 0, stream>>>(esrc, edst, dinv, x, p1, E, F);
  k_self<<<(t4a + 255) / 256, 256, 0, stream>>>((const float4*)p1, dinv, (float4*)p2, N, F / 4);
  k_spmm<<<(E + 7) / 8, 256, 0, stream>>>(esrc, edst, dinv, p1, p2, E, F);

  // ---- layer 1 GEMMs (+bias, ReLU) into concatenated h[N, 384]
  dim3 g1((N + 127) / 128, (H + 15) / 16);
  k_gemm_wmma<<<g1, 256, 0, stream>>>(x,  F, W1_0, H, b1 + 0 * H, h + 0 * H, H3, N, F, 1);
  k_gemm_wmma<<<g1, 256, 0, stream>>>(p1, F, W1_1, H, b1 + 1 * H, h + 1 * H, H3, N, F, 1);
  k_gemm_wmma<<<g1, 256, 0, stream>>>(p2, F, W1_2, H, b1 + 2 * H, h + 2 * H, H3, N, F, 1);

  // ---- layer 2 propagates: q1 = P(h), q2 = P(q1)
  int t4b = N * (H3 / 4);
  k_self<<<(t4b + 255) / 256, 256, 0, stream>>>((const float4*)h, dinv, (float4*)q1, N, H3 / 4);
  k_spmm<<<(E + 7) / 8, 256, 0, stream>>>(esrc, edst, dinv, h, q1, E, H3);
  k_self<<<(t4b + 255) / 256, 256, 0, stream>>>((const float4*)q1, dinv, (float4*)q2, N, H3 / 4);
  k_spmm<<<(E + 7) / 8, 256, 0, stream>>>(esrc, edst, dinv, q1, q2, E, H3);

  // ---- layer 2 GEMMs (+bias) into d_out[N, 120]
  dim3 g2((N + 127) / 128, (Cc + 15) / 16);
  k_gemm_wmma<<<g2, 256, 0, stream>>>(h,  H3, W2_0, Cc, b2 + 0 * Cc, outv + 0 * Cc, C3, N, H3, 0);
  k_gemm_wmma<<<g2, 256, 0, stream>>>(q1, H3, W2_1, Cc, b2 + 1 * Cc, outv + 1 * Cc, C3, N, H3, 0);
  k_gemm_wmma<<<g2, 256, 0, stream>>>(q2, H3, W2_2, Cc, b2 + 2 * Cc, outv + 2 * Cc, C3, N, H3, 0);

  // ---- fused log_softmax over 120 classes, in place
  k_logsoftmax<<<(N + 7) / 8, 256, 0, stream>>>(outv, N, C3);
}
